// WorldModel_47699906789935
// MI455X (gfx1250) — matrix operand order repaired and checked by reading, not compiled, not run on hardware
//
#include <hip/hip_runtime.h>
#include <hip/hip_bf16.h>

// ---------------------------------------------------------------------------
// Problem constants
// ---------------------------------------------------------------------------
#define NROWS  32768      // B*T = 1024*32
#define DIMIN  640
#define HID    512
#define DOUT   288        // sum(SPLITS)
#define NJ     24

typedef __attribute__((ext_vector_type(16))) _Float16 v16h;
typedef __attribute__((ext_vector_type(8)))  _Float16 h8;
typedef __attribute__((ext_vector_type(4)))  _Float16 h4;
typedef __attribute__((ext_vector_type(8)))  float    v8f;

// ---------------------------------------------------------------------------
// f32 -> f16 conversion, 4 elements per thread (all sizes divisible by 4)
// ---------------------------------------------------------------------------
__global__ __launch_bounds__(256) void cvt_f32_to_f16_x4(const float* __restrict__ src,
                                                         _Float16* __restrict__ dst, int n4) {
    int i = blockIdx.x * blockDim.x + threadIdx.x;
    if (i < n4) {
        float4 v = ((const float4*)src)[i];
        h4 o;
        o[0] = (_Float16)v.x; o[1] = (_Float16)v.y;
        o[2] = (_Float16)v.z; o[3] = (_Float16)v.w;
        ((h4*)dst)[i] = o;
    }
}

// ---------------------------------------------------------------------------
// Common WMMA fragment math (CDNA5 ISA 7.12.2):
//   A (16x32 f16): lanes 0-15 row M=lane, halves = K{0..15} (contiguous, code's order);
//                  lanes 16-31 row M=lane-16, halves = K{16..31}
//   B (32x16 f16): col N = lane%16; lanes 0-15 halves K{0..15}; 16-31 K{16..31} (code's order, same as A; not ISA table)
//   C/D (16x16 f32): col = lane%16; VGPR r -> row r (lanes 0-15) / r+8 (16-31)
// ---------------------------------------------------------------------------

// ---------------------------------------------------------------------------
// LDS-B WMMA GEMM (hidden layers): the whole block shares one B panel of
// NTB*16 columns staged once in LDS (NTB=4 -> 64 cols x K halfs = 64-80 KB).
// 8 waves span 16 M-tiles (256 rows) reading A straight from global.
//   L2 traffic/layer: B 0.5MB * (32768/256) = 67MB, A = 8 passes of act matrix
//   (was ~537MB + ~268MB with per-wave B loads).
// K-loop fully unrolled + register double-buffer (A from global, B from LDS).
// ---------------------------------------------------------------------------
template<int K, int OUT, int MT, int NTB, bool SILU, bool HALF_OUT>
__global__ __launch_bounds__(256) void gemm_wmma_ldsB(const _Float16* __restrict__ A,
                                                      const _Float16* __restrict__ W,
                                                      const float* __restrict__ bias,
                                                      void* __restrict__ outP)
{
    const int lane     = threadIdx.x & 31;
    const int wave     = threadIdx.x >> 5;
    const int nBlocksN = (OUT / 16) / NTB;
    const int bn       = blockIdx.x % nBlocksN;
    const int bm       = blockIdx.x / nBlocksN;
    const int tileN0   = bn * NTB;                 // block-wide N panel
    const int tileM0   = (bm * 8 + wave) * MT;     // per-wave M tiles

    __shared__ _Float16 sB[NTB * 16 * K];

    // cooperative stage of the contiguous W panel (rows tileN0*16 .. +NTB*16)
    {
        const h8* __restrict__ src = (const h8*)(W + (size_t)tileN0 * 16 * K);
        h8* dst = (h8*)sB;
        constexpr int nVec = NTB * 16 * K / 8;
        #pragma unroll
        for (int t = 0; t < nVec / 256; ++t)
            dst[t * 256 + threadIdx.x] = src[t * 256 + threadIdx.x];
    }
    __syncthreads();

    const int hi16  = lane >> 4;
    const int koffA = hi16 * 16;
    const int koffB = hi16 * 16;

    const _Float16* aRow[MT];
    #pragma unroll
    for (int mi = 0; mi < MT; ++mi)
        aRow[mi] = A + (size_t)((tileM0 + mi) * 16 + (lane & 15)) * K;
    const _Float16* bRow[NTB];
    #pragma unroll
    for (int ni = 0; ni < NTB; ++ni)
        bRow[ni] = sB + (size_t)(ni * 16 + (lane & 15)) * K;

    auto loadA = [&](v16h (&dst)[MT], int k0) {
        #pragma unroll
        for (int mi = 0; mi < MT; ++mi) {
            h8 lo = *(const h8*)(aRow[mi] + k0 + koffA);
            h8 hi = *(const h8*)(aRow[mi] + k0 + koffA + 8);
            dst[mi] = __builtin_shufflevector(lo, hi, 0,1,2,3,4,5,6,7,8,9,10,11,12,13,14,15);
        }
    };
    auto loadB = [&](v16h (&dst)[NTB], int k0) {
        #pragma unroll
        for (int ni = 0; ni < NTB; ++ni) {
            h8 lo = *(const h8*)(bRow[ni] + k0 + koffB);
            h8 hi = *(const h8*)(bRow[ni] + k0 + koffB + 8);
            dst[ni] = __builtin_shufflevector(lo, hi, 0,1,2,3,4,5,6,7,8,9,10,11,12,13,14,15);
        }
    };

    v8f c[MT][NTB];
    #pragma unroll
    for (int mi = 0; mi < MT; ++mi)
        #pragma unroll
        for (int ni = 0; ni < NTB; ++ni)
            c[mi][ni] = (v8f){};

    v16h a[MT], b[NTB];
    loadA(a, 0);
    loadB(b, 0);
    #pragma unroll
    for (int k0 = 0; k0 < K; k0 += 32) {
        v16h an[MT], bn2[NTB];
        const bool more = (k0 + 32 < K);   // compile-time after full unroll
        if (more) { loadA(an, k0 + 32); loadB(bn2, k0 + 32); }
        #pragma unroll
        for (int mi = 0; mi < MT; ++mi)
            #pragma unroll
            for (int ni = 0; ni < NTB; ++ni)
                c[mi][ni] = __builtin_amdgcn_wmma_f32_16x16x32_f16(
                    false, a[mi], false, b[ni], (short)0, c[mi][ni], false, false);
        if (more) {
            #pragma unroll
            for (int mi = 0; mi < MT; ++mi) a[mi] = an[mi];
            #pragma unroll
            for (int ni = 0; ni < NTB; ++ni) b[ni] = bn2[ni];
        }
    }

    #pragma unroll
    for (int ni = 0; ni < NTB; ++ni) {
        const int col = (tileN0 + ni) * 16 + (lane & 15);
        const float bv = bias[col];
        #pragma unroll
        for (int mi = 0; mi < MT; ++mi) {
            const int rowBase = (tileM0 + mi) * 16 + hi16 * 8;
            #pragma unroll
            for (int r = 0; r < 8; ++r) {
                float v = c[mi][ni][r] + bv;
                if (SILU) v = v * __builtin_amdgcn_rcpf(1.0f + __expf(-v));
                size_t idx = (size_t)(rowBase + r) * OUT + col;
                if (HALF_OUT) ((_Float16*)outP)[idx] = (_Float16)v;
                else          ((float*)outP)[idx]    = v;
            }
        }
    }
}

// ---------------------------------------------------------------------------
// Direct-global WMMA GEMM (head layer: 18 N-tiles, panel too big for LDS win)
// ---------------------------------------------------------------------------
template<int K, int OUT, int MT, int NT, bool SILU, bool HALF_OUT>
__global__ __launch_bounds__(256) void gemm_wmma(const _Float16* __restrict__ A,
                                                 const _Float16* __restrict__ W,
                                                 const float* __restrict__ bias,
                                                 void* __restrict__ outP)
{
    const int lane    = threadIdx.x & 31;
    const int wave    = threadIdx.x >> 5;
    const int nGroups = (OUT / 16) / NT;
    const int w       = blockIdx.x * 8 + wave;
    const int tileM0  = (w / nGroups) * MT;
    const int tileN0  = (w % nGroups) * NT;

    const int hi16  = lane >> 4;
    const int koffA = hi16 * 16;
    const int koffB = hi16 * 16;

    const _Float16* aRow[MT];
    const _Float16* bRow[NT];
    #pragma unroll
    for (int mi = 0; mi < MT; ++mi)
        aRow[mi] = A + (size_t)((tileM0 + mi) * 16 + (lane & 15)) * K;
    #pragma unroll
    for (int ni = 0; ni < NT; ++ni)
        bRow[ni] = W + (size_t)((tileN0 + ni) * 16 + (lane & 15)) * K;

    auto loadA = [&](v16h (&dst)[MT], int k0) {
        #pragma unroll
        for (int mi = 0; mi < MT; ++mi) {
            h8 lo = *(const h8*)(aRow[mi] + k0 + koffA);
            h8 hi = *(const h8*)(aRow[mi] + k0 + koffA + 8);
            dst[mi] = __builtin_shufflevector(lo, hi, 0,1,2,3,4,5,6,7,8,9,10,11,12,13,14,15);
        }
    };
    auto loadB = [&](v16h (&dst)[NT], int k0) {
        #pragma unroll
        for (int ni = 0; ni < NT; ++ni) {
            h8 lo = *(const h8*)(bRow[ni] + k0 + koffB);
            h8 hi = *(const h8*)(bRow[ni] + k0 + koffB + 8);
            dst[ni] = __builtin_shufflevector(lo, hi, 0,1,2,3,4,5,6,7,8,9,10,11,12,13,14,15);
        }
    };

    v8f c[MT][NT];
    #pragma unroll
    for (int mi = 0; mi < MT; ++mi)
        #pragma unroll
        for (int ni = 0; ni < NT; ++ni)
            c[mi][ni] = (v8f){};

    v16h a[MT], b[NT];
    loadA(a, 0);
    loadB(b, 0);
    #pragma unroll
    for (int k0 = 0; k0 < K; k0 += 32) {
        v16h an[MT], bn[NT];
        const bool more = (k0 + 32 < K);
        if (more) { loadA(an, k0 + 32); loadB(bn, k0 + 32); }
        #pragma unroll
        for (int mi = 0; mi < MT; ++mi)
            #pragma unroll
            for (int ni = 0; ni < NT; ++ni)
                c[mi][ni] = __builtin_amdgcn_wmma_f32_16x16x32_f16(
                    false, a[mi], false, b[ni], (short)0, c[mi][ni], false, false);
        if (more) {
            #pragma unroll
            for (int mi = 0; mi < MT; ++mi) a[mi] = an[mi];
            #pragma unroll
            for (int ni = 0; ni < NT; ++ni) b[ni] = bn[ni];
        }
    }

    #pragma unroll
    for (int ni = 0; ni < NT; ++ni) {
        const int col = (tileN0 + ni) * 16 + (lane & 15);
        const float bv = bias[col];
        #pragma unroll
        for (int mi = 0; mi < MT; ++mi) {
            const int rowBase = (tileM0 + mi) * 16 + hi16 * 8;
            #pragma unroll
            for (int r = 0; r < 8; ++r) {
                float v = c[mi][ni][r] + bv;
                if (SILU) v = v * __builtin_amdgcn_rcpf(1.0f + __expf(-v));
                size_t idx = (size_t)(rowBase + r) * OUT + col;
                if (HALF_OUT) ((_Float16*)outP)[idx] = (_Float16)v;
                else          ((float*)outP)[idx]    = v;
            }
        }
    }
}

// ---------------------------------------------------------------------------
// Quaternion helpers (match JAX reference, xyzw order)
// ---------------------------------------------------------------------------
struct Q  { float x, y, z, w; };
struct V3 { float x, y, z; };

__device__ __forceinline__ V3 v3(float x, float y, float z) { V3 r{x, y, z}; return r; }
__device__ __forceinline__ V3 addv(V3 a, V3 b) { return v3(a.x + b.x, a.y + b.y, a.z + b.z); }
__device__ __forceinline__ V3 scl(V3 a, float s) { return v3(a.x * s, a.y * s, a.z * s); }
__device__ __forceinline__ float dotv(V3 a, V3 b) { return a.x * b.x + a.y * b.y + a.z * b.z; }
__device__ __forceinline__ V3 crossv(V3 a, V3 b) {
    return v3(a.y * b.z - a.z * b.y, a.z * b.x - a.x * b.z, a.x * b.y - a.y * b.x);
}
__device__ __forceinline__ Q quat_mul(Q a, Q b) {
    Q r;
    r.x = a.w * b.x + a.x * b.w + a.y * b.z - a.z * b.y;
    r.y = a.w * b.y + a.y * b.w + a.z * b.x - a.x * b.z;
    r.z = a.w * b.z + a.z * b.w + a.x * b.y - a.y * b.x;
    r.w = a.w * b.w - a.x * b.x - a.y * b.y - a.z * b.z;
    return r;
}
__device__ __forceinline__ V3 quat_rotate(Q q, V3 v) {
    V3 qv = v3(q.x, q.y, q.z);
    float w = q.w;
    V3 a = scl(v, 2.f * w * w - 1.f);
    V3 b = scl(crossv(qv, v), 2.f * w);
    V3 c = scl(qv, 2.f * dotv(qv, v));
    return addv(addv(a, b), c);
}
__device__ __forceinline__ float norm_angle(float x) { return atan2f(sinf(x), cosf(x)); }

__device__ __forceinline__ Q exp_map_to_quat(V3 e) {
    float angle = sqrtf(dotv(e, e));
    bool  m    = angle > 1e-5f;
    float safe = m ? angle : 1.f;
    V3    axis = m ? scl(e, 1.f / safe) : v3(0.f, 0.f, 1.f);
    float ang  = m ? norm_angle(angle) : 0.f;
    float th = 0.5f * ang;
    float s = sinf(th);
    Q q; q.x = axis.x * s; q.y = axis.y * s; q.z = axis.z * s; q.w = cosf(th);
    return q;
}
__device__ __forceinline__ V3 quat_to_exp_map(Q q) {
    float qw = fminf(fmaxf(q.w, -1.f + 1e-7f), 1.f - 1e-7f);
    float sin_t = sqrtf(1.f - qw * qw);
    bool  m    = sin_t > 1e-5f;
    float safe = m ? sin_t : 1.f;
    V3    axis = m ? v3(q.x / safe, q.y / safe, q.z / safe) : v3(0.f, 0.f, 1.f);
    float ang  = m ? norm_angle(2.f * acosf(qw)) : 0.f;
    return scl(axis, ang);
}

// ---------------------------------------------------------------------------
// Per-row kinematics postprocess.  One thread per row (N = 32768).
// ---------------------------------------------------------------------------
__global__ __launch_bounds__(128) void postprocess(
    const float* __restrict__ out3,        // [N, 288] MLP head output
    const float* __restrict__ rg_pos,      // [N, 72]
    const float* __restrict__ rg_rot,      // [N, 96]
    const float* __restrict__ dof_pos,     // [N, 69]
    const float* __restrict__ local_trans, // [24, 3]
    float* __restrict__ out)
{
    const int n = blockIdx.x * blockDim.x + threadIdx.x;
    if (n >= NROWS) return;

    constexpr size_t OFF_BP = 0;
    constexpr size_t OFF_BQ = (size_t)NROWS * 72;
    constexpr size_t OFF_DP = OFF_BQ + (size_t)NROWS * 96;
    constexpr size_t OFF_LV = OFF_DP + (size_t)NROWS * 69;
    constexpr size_t OFF_AV = OFF_LV + (size_t)NROWS * 72;
    constexpr size_t OFF_DV = OFF_AV + (size_t)NROWS * 72;

    const float* __restrict__ o = out3 + (size_t)n * DOUT;
    V3 pos_delta = v3(o[0], o[1], o[2]);
    V3 rot_delta = v3(o[3], o[4], o[5]);

    V3 root_pos = v3(rg_pos[(size_t)n * 72 + 0], rg_pos[(size_t)n * 72 + 1], rg_pos[(size_t)n * 72 + 2]);
    Q  root_rot;
    root_rot.x = rg_rot[(size_t)n * 96 + 0];
    root_rot.y = rg_rot[(size_t)n * 96 + 1];
    root_rot.z = rg_rot[(size_t)n * 96 + 2];
    root_rot.w = rg_rot[(size_t)n * 96 + 3];

    V3 rdir = quat_rotate(root_rot, v3(1.f, 0.f, 0.f));
    float h = atan2f(rdir.y, rdir.x);
    float sh = sinf(0.5f * h), ch = cosf(0.5f * h);
    Q h_rot; h_rot.x = 0.f; h_rot.y = 0.f; h_rot.z =  sh; h_rot.w = ch;
    Q h_inv; h_inv.x = 0.f; h_inv.y = 0.f; h_inv.z = -sh; h_inv.w = ch;

    V3 new_root_pos = addv(root_pos, quat_rotate(h_rot, pos_delta));
    Q  rd = quat_mul(quat_mul(h_rot, exp_map_to_quat(rot_delta)), h_inv);
    Q  new_root_rot = quat_mul(rd, root_rot);

    Q ndq[23];
    #pragma unroll
    for (int j = 0; j < 23; ++j) {
        V3 dq = v3(dof_pos[(size_t)n * 69 + 3 * j + 0],
                   dof_pos[(size_t)n * 69 + 3 * j + 1],
                   dof_pos[(size_t)n * 69 + 3 * j + 2]);
        V3 dd = v3(o[6 + 3 * j + 0], o[6 + 3 * j + 1], o[6 + 3 * j + 2]);
        Q cur = exp_map_to_quat(dq);
        Q del = exp_map_to_quat(dd);
        ndq[j] = quat_mul(del, cur);
        V3 em = quat_to_exp_map(ndq[j]);
        out[OFF_DP + (size_t)n * 69 + 3 * j + 0] = em.x;
        out[OFF_DP + (size_t)n * 69 + 3 * j + 1] = em.y;
        out[OFF_DP + (size_t)n * 69 + 3 * j + 2] = em.z;
    }

    const int par[NJ] = {-1, 0, 0, 0, 1, 2, 3, 4, 5, 6, 7, 8, 9, 9, 9, 12, 13, 14, 16, 17, 18, 19, 20, 21};
    float px[NJ], py[NJ], pz[NJ];
    Q rw[NJ];
    px[0] = new_root_pos.x; py[0] = new_root_pos.y; pz[0] = new_root_pos.z;
    rw[0] = new_root_rot;
    #pragma unroll
    for (int i = 1; i < NJ; ++i) {
        int p = par[i];
        V3 lt = v3(local_trans[3 * i + 0], local_trans[3 * i + 1], local_trans[3 * i + 2]);
        V3 jp = addv(quat_rotate(rw[p], lt), v3(px[p], py[p], pz[p]));
        rw[i] = quat_mul(rw[p], ndq[i - 1]);
        px[i] = jp.x; py[i] = jp.y; pz[i] = jp.z;
    }
    #pragma unroll
    for (int i = 0; i < NJ; ++i) {
        out[OFF_BP + (size_t)n * 72 + 3 * i + 0] = px[i];
        out[OFF_BP + (size_t)n * 72 + 3 * i + 1] = py[i];
        out[OFF_BP + (size_t)n * 72 + 3 * i + 2] = pz[i];
        out[OFF_BQ + (size_t)n * 96 + 4 * i + 0] = rw[i].x;
        out[OFF_BQ + (size_t)n * 96 + 4 * i + 1] = rw[i].y;
        out[OFF_BQ + (size_t)n * 96 + 4 * i + 2] = rw[i].z;
        out[OFF_BQ + (size_t)n * 96 + 4 * i + 3] = rw[i].w;
    }

    #pragma unroll
    for (int i = 0; i < NJ; ++i) {
        V3 lv = v3(o[75 + 3 * i + 0], o[75 + 3 * i + 1], o[75 + 3 * i + 2]);
        V3 av = v3(o[147 + 3 * i + 0], o[147 + 3 * i + 1], o[147 + 3 * i + 2]);
        V3 rlv = quat_rotate(h_rot, lv);
        V3 rav = quat_rotate(h_rot, av);
        out[OFF_LV + (size_t)n * 72 + 3 * i + 0] = rlv.x;
        out[OFF_LV + (size_t)n * 72 + 3 * i + 1] = rlv.y;
        out[OFF_LV + (size_t)n * 72 + 3 * i + 2] = rlv.z;
        out[OFF_AV + (size_t)n * 72 + 3 * i + 0] = rav.x;
        out[OFF_AV + (size_t)n * 72 + 3 * i + 1] = rav.y;
        out[OFF_AV + (size_t)n * 72 + 3 * i + 2] = rav.z;
    }
    #pragma unroll
    for (int k = 0; k < 69; ++k)
        out[OFF_DV + (size_t)n * 69 + k] = o[219 + k];
}

// ---------------------------------------------------------------------------
// Launch
// ---------------------------------------------------------------------------
extern "C" void kernel_launch(void* const* d_in, const int* in_sizes, int n_in,
                              void* d_out, int out_size, void* d_ws, size_t ws_size,
                              hipStream_t stream)
{
    const float* x           = (const float*)d_in[0];
    const float* rg_pos      = (const float*)d_in[1];
    const float* rg_rot      = (const float*)d_in[2];
    const float* dof_pos     = (const float*)d_in[5];
    const float* W0          = (const float*)d_in[6];
    const float* b0          = (const float*)d_in[7];
    const float* W1          = (const float*)d_in[8];
    const float* b1          = (const float*)d_in[9];
    const float* W2          = (const float*)d_in[10];
    const float* b2          = (const float*)d_in[11];
    const float* W3          = (const float*)d_in[12];
    const float* b3          = (const float*)d_in[13];
    const float* local_trans = (const float*)d_in[14];

    char* ws = (char*)d_ws;
    size_t off = 0;
    _Float16* xh  = (_Float16*)(ws + off); off += (size_t)NROWS * DIMIN * 2;
    _Float16* h1  = (_Float16*)(ws + off); off += (size_t)NROWS * HID * 2;
    _Float16* h2  = (_Float16*)(ws + off); off += (size_t)NROWS * HID * 2;
    float*    o3  = (float*)   (ws + off); off += (size_t)NROWS * DOUT * 4;
    _Float16* w0h = (_Float16*)(ws + off); off += (size_t)HID * DIMIN * 2;
    _Float16* w1h = (_Float16*)(ws + off); off += (size_t)HID * HID * 2;
    _Float16* w2h = (_Float16*)(ws + off); off += (size_t)HID * HID * 2;
    _Float16* w3h = (_Float16*)(ws + off); off += (size_t)DOUT * HID * 2;

    {
        int n4 = (NROWS * DIMIN) / 4;
        cvt_f32_to_f16_x4<<<(n4 + 255) / 256, 256, 0, stream>>>(x, xh, n4);
        n4 = (HID * DIMIN) / 4;
        cvt_f32_to_f16_x4<<<(n4 + 255) / 256, 256, 0, stream>>>(W0, w0h, n4);
        n4 = (HID * HID) / 4;
        cvt_f32_to_f16_x4<<<(n4 + 255) / 256, 256, 0, stream>>>(W1, w1h, n4);
        cvt_f32_to_f16_x4<<<(n4 + 255) / 256, 256, 0, stream>>>(W2, w2h, n4);
        n4 = (DOUT * HID) / 4;
        cvt_f32_to_f16_x4<<<(n4 + 255) / 256, 256, 0, stream>>>(W3, w3h, n4);
    }

    // Hidden layers: LDS-B GEMM. Block = 8 waves x (MT=2 M-tiles) = 256 rows,
    // N panel = NTB=4 tiles (64 cols) staged in LDS.
    // grid = (2048/16 M-groups) * (32/4 N-groups) = 128*8 = 1024 blocks
    const int blocksHid  = ((NROWS / 16) / (8 * 2)) * ((HID / 16) / 4);   // 1024
    // Head layer: direct-global GEMM, MT=2, NT=3 -> 768 blocks
    const int blocksHead = ((NROWS / 16) / 2) * ((DOUT / 16) / 3) / 8;    // 768
    gemm_wmma_ldsB<DIMIN, HID, 2, 4, true, true ><<<blocksHid,  256, 0, stream>>>(xh, w0h, b0, h1);
    gemm_wmma_ldsB<HID,   HID, 2, 4, true, true ><<<blocksHid,  256, 0, stream>>>(h1, w1h, b1, h2);
    gemm_wmma_ldsB<HID,   HID, 2, 4, true, true ><<<blocksHid,  256, 0, stream>>>(h2, w2h, b2, h1);
    gemm_wmma<HID, DOUT, 2, 3, false, false>     <<<blocksHead, 256, 0, stream>>>(h1, w3h, b3, o3);

    postprocess<<<NROWS / 128, 128, 0, stream>>>(o3, rg_pos, rg_rot, dof_pos, local_trans,
                                                 (float*)d_out);
}
